// SPGG_85504208929337
// MI455X (gfx1250) — matrix operand order repaired and checked by reading, not compiled, not run on hardware
//
#include <hip/hip_runtime.h>
#include <math.h>

// ---------------------------------------------------------------------------
// SPGG step on 4096x4096 periodic lattice, CDNA5 (gfx1250, wave32).
// Stencil chain S1=K s, S2=K^2 s, S3=K^3 s computed as banded bf16 matmuls on
// the WMMA pipe (exact: s is {0,1}, all intermediates are small integers).
//   personal = (r/5)(S2+S1) - 6 s
//   team     = (r/25)(S3+S2) - (6/5) S1
//   combined = (1-w) personal + w team,  w = sigmoid(beta * mean_tu/(mean_pr+eps))
//   encoded  = interleaved [s, S1, mean(s)]
// B fragments (banded weight matrices) are lane-only functions: built once per
// workgroup into LDS, reused by every tile -> no per-tile VALU rebuild.
// ---------------------------------------------------------------------------

#define L_DIM 4096
#define NCELL ((size_t)L_DIM * (size_t)L_DIM)

typedef __attribute__((ext_vector_type(16))) __bf16 v16bf;
typedef __attribute__((ext_vector_type(8)))  float  v8f;

__device__ __forceinline__ unsigned short f2bf(float f) {
  // truncation is exact for {0,1} and for integer weights <= 13
  return (unsigned short)(__builtin_bit_cast(unsigned int, f) >> 16);
}
__device__ __forceinline__ float bf2f(unsigned short b) {
  return __builtin_bit_cast(float, (unsigned int)b << 16);
}

// Nibble-packed diamond-conv weight rows for K, K^2, K^3 (dx = -3..3).
// Fragment order: f 0..6 = K^3 rows d=-3..3, f 7..11 = K^2 d=-2..2,
//                 f 12..14 = K d=-1..1.
__device__ const unsigned int WROWS[15] = {
    0x1000u, 0x33300u, 0x36C630u, 0x13CDC31u, 0x36C630u, 0x33300u, 0x1000u,
    0x1000u, 0x22200u, 0x125210u, 0x22200u,  0x1000u,
    0x1000u, 0x11100u, 0x1000u};

// A fragment (16x32 bf16) from a 64-byte LDS row. A layout used here (wave32, contiguous k):
// lane holds row M=lane%16; elements 0..15 -> K = 16g..16g+15
__device__ __forceinline__ v16bf load_a(const unsigned short* rowp, int g) {
  union { v16bf v; uint4 q[2]; } a;
  const uint4* p = (const uint4*)rowp;  // 4 x 16B chunks of 8 halfs each
  a.q[0] = p[2 * g];
  a.q[1] = p[2 * g + 1];
  return a.v;
}

// Read a prebuilt 32-byte per-lane B fragment from LDS.
__device__ __forceinline__ v16bf load_bfrag(const unsigned short* p) {
  union { v16bf v; uint4 q[2]; } b;
  const uint4* q = (const uint4*)p;
  b.q[0] = q[0];
  b.q[1] = q[1];
  return b.v;
}

__global__ void spgg_init_kernel(float* acc) {
  if (threadIdx.x < 8) acc[threadIdx.x] = 0.0f;
}

// One 16x16 output tile per wave; 8 waves (8 tiles) per 256-thread workgroup.
__global__ void __launch_bounds__(256) spgg_stencil_kernel(
    const float* __restrict__ s, const float* __restrict__ rp,
    float* __restrict__ out, float* __restrict__ acc) {
  __shared__ __align__(16) unsigned short tile[8][22][32];   // halo-3 rows, halo-8 cols
  __shared__ __align__(16) unsigned short bfrag[15][32][16]; // 15 banded B fragments

  const int lane    = threadIdx.x & 31;
  const int w       = threadIdx.x >> 5;
  const int tileIdx = blockIdx.x * 8 + w;
  const int tx = tileIdx & 255;
  const int ty = tileIdx >> 8;
  const int row0 = ty * 16, col0 = tx * 16;

  const int j = lane & 15;   // A row index / B,C column index
  const int g = lane >> 4;

  // --- Build B fragments once per block (branchless, ~2 fragments per wave).
  // B layout used here (wave32, contiguous k, same as A): lane group g, column j; element e holds K=e+16g.
  // B[K][j] = w_row[K - 8 - j].
  for (int f = w; f < 15; f += 8) {
    const unsigned int row = WROWS[f];
    union { uint4 q[2]; unsigned short u[16]; } b;
#pragma unroll
    for (int e = 0; e < 16; ++e) {
      const int dx = e + 16 * g - 8 - j;          // column offset of this K
      const unsigned sh  = (unsigned)(4 * (dx + 3));
      const unsigned nib = (sh < 32u) ? ((row >> (sh & 31u)) & 0xFu) : 0u;
      b.u[e] = (unsigned short)(__builtin_bit_cast(unsigned int, (float)nib) >> 16);
    }
    uint4* dst = (uint4*)&bfrag[f][lane][0];
    dst[0] = b.q[0];
    dst[1] = b.q[1];
  }

  // --- Stage s tile: rows row0-3..row0+18, cols col0-8..col0+23, periodic wrap.
  const int gcol = (col0 - 8 + lane) & (L_DIM - 1);
  int grow = (row0 - 3) & (L_DIM - 1);
#pragma unroll
  for (int rr = 0; rr < 22; ++rr) {
    tile[w][rr][lane] = f2bf(s[(size_t)grow * L_DIM + gcol]);
    grow = (grow + 1) & (L_DIM - 1);
  }
  __syncthreads();

  // --- 15 banded WMMAs: accumulate S3 (7 rows), S2 (5 rows), S1 (3 rows).
  v8f c1 = {}; v8f c2 = {}; v8f c3 = {};
#pragma unroll
  for (int d = -3; d <= 3; ++d) {
    const v16bf aF = load_a(&tile[w][3 + j + d][0], g);
    c3 = __builtin_amdgcn_wmma_f32_16x16x32_bf16(
        false, aF, false, load_bfrag(&bfrag[d + 3][lane][0]), (short)0, c3,
        false, false);
    if (d >= -2 && d <= 2)
      c2 = __builtin_amdgcn_wmma_f32_16x16x32_bf16(
          false, aF, false, load_bfrag(&bfrag[9 + d][lane][0]), (short)0, c2,
          false, false);
    if (d >= -1 && d <= 1)
      c1 = __builtin_amdgcn_wmma_f32_16x16x32_bf16(
          false, aF, false, load_bfrag(&bfrag[13 + d][lane][0]), (short)0, c1,
          false, false);
  }

  const float r  = rp[0];
  const float r5 = r * 0.2f;

  const size_t base = (size_t)(row0 + 8 * g) * L_DIM + (size_t)(col0 + j);
  float sumS = 0.f, sumPr = 0.f, sumTu = 0.f;
#pragma unroll
  for (int k = 0; k < 8; ++k) {
    const float sc = bf2f(tile[w][3 + k + 8 * g][8 + j]);  // center s
    const float pr = r5 * (c2[k] + c1[k]) - 6.0f * sc;
    const float tu = (r5 * 0.2f) * (c3[k] + c2[k]) - 1.2f * c1[k];
    const size_t idx = base + (size_t)k * L_DIM;
    out[NCELL + idx]     = pr;
    out[2 * NCELL + idx] = tu;
    sumS += sc; sumPr += pr; sumTu += tu;
  }

  // wave32 butterfly reduce, three f32 global atomics per wave
#pragma unroll
  for (int m = 16; m >= 1; m >>= 1) {
    sumS  += __shfl_xor(sumS,  m, 32);
    sumPr += __shfl_xor(sumPr, m, 32);
    sumTu += __shfl_xor(sumTu, m, 32);
  }
  if (lane == 0) {
    atomicAdd(&acc[0], sumS);
    atomicAdd(&acc[1], sumPr);
    atomicAdd(&acc[2], sumTu);
  }
}

__global__ void spgg_scalar_kernel(const float* beta, float* acc) {
  const float invN   = 1.0f / (float)NCELL;
  const float meanPr = acc[1] * invN;
  const float meanTu = acc[2] * invN;
  const float ratio  = meanTu / (meanPr + 1e-8f);
  acc[3] = 1.0f / (1.0f + __expf(-beta[0] * ratio));  // w_t
  acc[4] = acc[0] * invN;                             // global cooperation
}

__global__ void __launch_bounds__(256) spgg_final_kernel(
    const float* __restrict__ s, const float* __restrict__ acc,
    float* __restrict__ out) {
  const size_t i   = (size_t)blockIdx.x * 256 + threadIdx.x;
  const int    row = (int)(i >> 12);
  const int    col = (int)(i & (L_DIM - 1));
  const float  wT  = acc[3];
  const float  gc  = acc[4];

  const size_t up = (size_t)((row - 1) & (L_DIM - 1)) * L_DIM + col;
  const size_t dn = (size_t)((row + 1) & (L_DIM - 1)) * L_DIM + col;
  const size_t lf = (size_t)row * L_DIM + ((col - 1) & (L_DIM - 1));
  const size_t rt = (size_t)row * L_DIM + ((col + 1) & (L_DIM - 1));

  const float sc = s[i];
  const float nc = sc + s[up] + s[dn] + s[lf] + s[rt];   // neighbor_coop (halo hits L2)
  const float pr = out[NCELL + i];
  const float tu = out[2 * NCELL + i];

  out[i] = (1.0f - wT) * pr + wT * tu;                   // combined_reward

  float* enc = out + 3 * NCELL + 3 * i;                  // interleaved [s, nc, gc]
  enc[0] = sc;
  enc[1] = nc;
  enc[2] = gc;
}

extern "C" void kernel_launch(void* const* d_in, const int* in_sizes, int n_in,
                              void* d_out, int out_size, void* d_ws, size_t ws_size,
                              hipStream_t stream) {
  const float* s    = (const float*)d_in[0];
  const float* r    = (const float*)d_in[1];
  const float* beta = (const float*)d_in[2];
  float* out = (float*)d_out;
  float* acc = (float*)d_ws;   // [sumS, sumPr, sumTu, w_t, gc, pad...]

  (void)in_sizes; (void)n_in; (void)out_size; (void)ws_size;

  spgg_init_kernel<<<1, 32, 0, stream>>>(acc);
  // 256x256 tiles of 16x16, 8 tiles (waves) per 256-thread block
  spgg_stencil_kernel<<<(256 * 256) / 8, 256, 0, stream>>>(s, r, out, acc);
  spgg_scalar_kernel<<<1, 1, 0, stream>>>(beta, acc);
  spgg_final_kernel<<<(unsigned)(NCELL / 256), 256, 0, stream>>>(s, acc, out);
}
